// GraphAttentionLayer_71244917506562
// MI455X (gfx1250) — matrix operand order repaired and checked by reading, not compiled, not run on hardware
//
#include <hip/hip_runtime.h>

// ---------------------------------------------------------------------------
// GraphAttentionLayer for MI455X (gfx1250, wave32, WMMA)
// Flash-attention (64 keys/iter) with v_wmma_f32_16x16x32_bf16, DPP row
// reductions, async global->LDS staging when available, LDS-transposed V
// tile (b128 fragment loads), all-bf16 projection GEMM + fused BatchNorm
// stats, normalize+SELU epilogue.
// Workspace (~10.2 MB): qbf(2M) kbf(2M) aggbf(2M) pre(4M) stats wbfA wbfR
// ---------------------------------------------------------------------------

#define BATCH   4
#define NSEQ    2048
#define DIM     128
#define TILE_K  64            // keys processed per iteration
#define WAVES   4             // waves per block; each wave owns 16 rows
#define ROWS_PB (WAVES * 16)  // 64 rows per block

#define SELU_ALPHA 1.6732632423543772f
#define SELU_SCALE 1.0507009873554805f

#if defined(__has_builtin)
#  if __has_builtin(__builtin_amdgcn_global_load_async_to_lds_b128) && \
      __has_builtin(__builtin_amdgcn_s_wait_asynccnt)
#    define USE_ASYNC_COPY 1
#  endif
#  if __has_builtin(__builtin_amdgcn_update_dpp)
#    define USE_DPP 1
#  endif
#endif

typedef __attribute__((ext_vector_type(16))) __bf16 v16bf;
typedef __attribute__((ext_vector_type(8)))  float  v8f;
typedef int v4i __attribute__((vector_size(16)));
typedef __attribute__((address_space(1))) v4i* gptr_b128;   // global b128 ptr
typedef __attribute__((address_space(3))) v4i* sptr_b128;   // LDS b128 ptr

union FragBF {
    v16bf v;
    uint4 q[2];              // q[0] = elements 0..7, q[1] = elements 8..15
    unsigned short u[16];
};

__device__ __forceinline__ unsigned short f2bf(float f) {
    unsigned int u = __float_as_uint(f);
    u += 0x7FFFu + ((u >> 16) & 1u);   // round-to-nearest-even
    return (unsigned short)(u >> 16);
}

// 16-lane (DPP row) reduce + broadcast: pure VALU, no LDS unit traffic.
#ifdef USE_DPP
__device__ __forceinline__ float rowmax16(float x) {
    x = fmaxf(x, __int_as_float(__builtin_amdgcn_update_dpp(0, __float_as_int(x), 0x128, 0xf, 0xf, true))); // row_ror:8
    x = fmaxf(x, __int_as_float(__builtin_amdgcn_update_dpp(0, __float_as_int(x), 0x124, 0xf, 0xf, true))); // row_ror:4
    x = fmaxf(x, __int_as_float(__builtin_amdgcn_update_dpp(0, __float_as_int(x), 0x122, 0xf, 0xf, true))); // row_ror:2
    x = fmaxf(x, __int_as_float(__builtin_amdgcn_update_dpp(0, __float_as_int(x), 0x121, 0xf, 0xf, true))); // row_ror:1
    return x;
}
__device__ __forceinline__ float rowsum16(float x) {
    x += __int_as_float(__builtin_amdgcn_update_dpp(0, __float_as_int(x), 0x128, 0xf, 0xf, true));
    x += __int_as_float(__builtin_amdgcn_update_dpp(0, __float_as_int(x), 0x124, 0xf, 0xf, true));
    x += __int_as_float(__builtin_amdgcn_update_dpp(0, __float_as_int(x), 0x122, 0xf, 0xf, true));
    x += __int_as_float(__builtin_amdgcn_update_dpp(0, __float_as_int(x), 0x121, 0xf, 0xf, true));
    return x;
}
#else
__device__ __forceinline__ float rowmax16(float x) {
    for (int m = 1; m <= 8; m <<= 1) x = fmaxf(x, __shfl_xor(x, m, 32));
    return x;
}
__device__ __forceinline__ float rowsum16(float x) {
    for (int m = 1; m <= 8; m <<= 1) x += __shfl_xor(x, m, 32);
    return x;
}
#endif

// ---------------------------------------------------------------------------
// Kernel 1: prep — qbf = bf16(x*w_map), kbf = bf16(x), bf16 weights,
//           zero BN accumulators (must be re-zeroed every call).
// ---------------------------------------------------------------------------
__global__ __launch_bounds__(256) void gal_prep(
    const float* __restrict__ x, const float* __restrict__ wmap,
    const float* __restrict__ Watt, const float* __restrict__ Wres,
    unsigned short* __restrict__ qbf, unsigned short* __restrict__ kbf,
    unsigned short* __restrict__ wbfA, unsigned short* __restrict__ wbfR,
    float* __restrict__ stats)
{
    const int idx = blockIdx.x * 256 + threadIdx.x;
    if (idx < 256) stats[idx] = 0.0f;                       // sum / sumsq
    if (idx < DIM * DIM) wbfA[idx] = f2bf(Watt[idx]);
    else if (idx < 2 * DIM * DIM) wbfR[idx - DIM * DIM] = f2bf(Wres[idx - DIM * DIM]);
    const int d = idx & (DIM - 1);
    const float xv = x[idx];
    kbf[idx] = f2bf(xv);
    qbf[idx] = f2bf(xv * wmap[d]);
}

// ---------------------------------------------------------------------------
// Kernel 2: flash attention.  aggbf[b,n,:] = bf16(softmax_m(q_n.k_m) @ x)
// One wave = 16-row tile; block stages a 64x128 bf16 K/V tile in LDS and a
// transposed copy for b128 V-fragment loads.
// ---------------------------------------------------------------------------
__global__ __launch_bounds__(128) void gal_attn(
    const unsigned short* __restrict__ qbf,
    const unsigned short* __restrict__ kbf,
    unsigned short* __restrict__ aggbf)
{
    __shared__ alignas(16) unsigned short ktile[TILE_K][DIM];        // 16 KB
    __shared__ alignas(16) unsigned short vtile[DIM][TILE_K];        // 16 KB (transposed)
    __shared__ alignas(16) unsigned short ptile[WAVES][16][TILE_K];  //  8 KB

    const int lane = threadIdx.x & 31;
    const int wave = threadIdx.x >> 5;
    const int b    = blockIdx.x / (NSEQ / ROWS_PB);
    const int m0   = (blockIdx.x % (NSEQ / ROWS_PB)) * ROWS_PB + wave * 16;

    const int nidx = lane & 15;        // A-row / B-col / C-col index
    const int khi  = (lane >> 4) & 1;  // lane-half selector

    // transpose work assignment: each thread owns half a key-row
    const int tkey = threadIdx.x >> 1;          // 0..63
    const int tdb  = (threadIdx.x & 1) * 64;    // dim base 0 or 64

    // Q A-fragments: 4 chunks of K=32 covering d=0..127 (wave-resident)
    FragBF qf[4];
    {
        const unsigned short* qb = qbf + ((size_t)b * NSEQ + (m0 + nidx)) * DIM;
        #pragma unroll
        for (int c = 0; c < 4; ++c) {
            qf[c].q[0] = *(const uint4*)(qb + c * 32 + khi * 8);       // K = c*32+khi*8+0..7
            qf[c].q[1] = *(const uint4*)(qb + c * 32 + 16 + khi * 8);  // K = +16
        }
    }

    v8f   o[8] = {};                   // O accumulator, 16x128 f32, C-layout
    float mrun[8], lrun[8];
    #pragma unroll
    for (int e = 0; e < 8; ++e) { mrun[e] = -1e30f; lrun[e] = 0.0f; }

    for (int k0 = 0; k0 < NSEQ; k0 += TILE_K) {
        __syncthreads();               // previous tile fully consumed
        // ---- stage 64x128 bf16 K/V tile (1024 x b128) --------------------
        {
            const unsigned short* src = kbf + ((size_t)b * NSEQ + k0) * DIM;
#ifdef USE_ASYNC_COPY
            #pragma unroll
            for (int t = 0; t < 8; ++t) {
                const unsigned off = (threadIdx.x + t * 128) * 8;   // ushort index
                __builtin_amdgcn_global_load_async_to_lds_b128(
                    (gptr_b128)(src + off),
                    (sptr_b128)(&ktile[0][0] + off),
                    0, 0);
            }
            __builtin_amdgcn_s_wait_asynccnt(0);
#else
            const uint4* s4 = (const uint4*)src;
            uint4* d4 = (uint4*)&ktile[0][0];
            #pragma unroll
            for (int t = 0; t < 8; ++t) d4[threadIdx.x + t * 128] = s4[threadIdx.x + t * 128];
#endif
        }
        if (k0 + TILE_K < NSEQ)        // global_prefetch_b8 next tile
            __builtin_prefetch((const void*)(kbf + ((size_t)b * NSEQ + k0 + TILE_K) * DIM +
                                             (size_t)threadIdx.x * 64), 0, 1);
        __syncthreads();

        // ---- build transposed V tile (reads ktile, writes vtile) ---------
        // (ordered vs PV-phase reads by the ptile barrier below)
        #pragma unroll
        for (int j = 0; j < 8; ++j) {
            uint4 rowq = *(const uint4*)&ktile[tkey][tdb + j * 8];
            const unsigned short* rw = (const unsigned short*)&rowq;
            #pragma unroll
            for (int u = 0; u < 8; ++u)
                vtile[tdb + j * 8 + u][tkey] = rw[u];
        }

        // ---- S = Q (16x128) . K^T (128x64) : four 16x16 C-tiles ----------
        v8f s[4];
        #pragma unroll
        for (int t = 0; t < 4; ++t) {
            FragBF kf[4];              // load all chunks first -> pipelined
            #pragma unroll
            for (int c = 0; c < 4; ++c) {
                const unsigned short* kb = &ktile[t * 16 + nidx][c * 32 + khi * 8];
                kf[c].q[0] = *(const uint4*)(kb);
                kf[c].q[1] = *(const uint4*)(kb + 16);
            }
            v8f acc = {};
            #pragma unroll
            for (int c = 0; c < 4; ++c)
                acc = __builtin_amdgcn_wmma_f32_16x16x32_bf16(
                        false, qf[c].v, false, kf[c].v, (short)0, acc, false, false);
            s[t] = acc;
        }

        // ---- online softmax (row = e + khi*8 lives in one 16-lane half) --
        float scl[8], p[4][8];
        #pragma unroll
        for (int e = 0; e < 8; ++e) {
            float t = fmaxf(fmaxf(s[0][e], s[1][e]), fmaxf(s[2][e], s[3][e]));
            t = rowmax16(t);
            const float mnew = fmaxf(mrun[e], t);
            scl[e] = __expf(mrun[e] - mnew);
            float rs = 0.0f;
            #pragma unroll
            for (int tt = 0; tt < 4; ++tt) { p[tt][e] = __expf(s[tt][e] - mnew); rs += p[tt][e]; }
            rs = rowsum16(rs);
            lrun[e] = lrun[e] * scl[e] + rs;
            mrun[e] = mnew;
        }
        #pragma unroll
        for (int dt = 0; dt < 8; ++dt)
            #pragma unroll
            for (int e = 0; e < 8; ++e) o[dt][e] = o[dt][e] * scl[e];

        // ---- transpose P (C-layout -> A-layout) via per-wave LDS ---------
        #pragma unroll
        for (int e = 0; e < 8; ++e)
            #pragma unroll
            for (int tt = 0; tt < 4; ++tt)
                ptile[wave][e + khi * 8][tt * 16 + nidx] = f2bf(p[tt][e]);
        __syncthreads();               // orders ptile AND vtile writes

        FragBF pf[2];   // A-frags: M=nidx, K(key)=kc*32+(i>>3)*16+khi*8+(i&7)
        #pragma unroll
        for (int kc = 0; kc < 2; ++kc) {
            pf[kc].q[0] = *(const uint4*)&ptile[wave][nidx][kc * 32 + khi * 8];
            pf[kc].q[1] = *(const uint4*)&ptile[wave][nidx][kc * 32 + 16 + khi * 8];
        }

        // ---- O += P (16x64) . V (64x128), 8 column tiles x 2 K-chunks ----
        #pragma unroll
        for (int dt = 0; dt < 8; ++dt) {
            FragBF vf[2];              // b128 loads from transposed tile
            #pragma unroll
            for (int kc = 0; kc < 2; ++kc) {
                const unsigned short* vb = &vtile[dt * 16 + nidx][kc * 32 + khi * 8];
                vf[kc].q[0] = *(const uint4*)(vb);
                vf[kc].q[1] = *(const uint4*)(vb + 16);
            }
            #pragma unroll
            for (int kc = 0; kc < 2; ++kc)
                o[dt] = __builtin_amdgcn_wmma_f32_16x16x32_bf16(
                            false, pf[kc].v, false, vf[kc].v, (short)0, o[dt], false, false);
        }
    }

    // ---- epilogue: O /= l, store agg as bf16 ------------------------------
    float inv[8];
    #pragma unroll
    for (int e = 0; e < 8; ++e) inv[e] = 1.0f / lrun[e];
    unsigned short* ab = aggbf + ((size_t)b * NSEQ + m0) * DIM;
    #pragma unroll
    for (int dt = 0; dt < 8; ++dt)
        #pragma unroll
        for (int e = 0; e < 8; ++e)
            ab[(size_t)(e + khi * 8) * DIM + dt * 16 + nidx] = f2bf(o[dt][e] * inv[e]);
}

// ---------------------------------------------------------------------------
// Kernel 3: pre = agg @ W_att^T + x @ W_res^T  (all-bf16 operands, f32 acc)
//           + per-channel BN partial sums (half-wave fold -> atomics)
// ---------------------------------------------------------------------------
__global__ __launch_bounds__(128) void gal_proj(
    const unsigned short* __restrict__ aggbf, const unsigned short* __restrict__ xbf,
    const unsigned short* __restrict__ wbfA, const unsigned short* __restrict__ wbfR,
    float* __restrict__ pre, float* __restrict__ stats)
{
    const int lane = threadIdx.x & 31;
    const int wave = threadIdx.x >> 5;
    const int r0   = blockIdx.x * ROWS_PB + wave * 16;   // over B*N = 8192 rows
    const int nidx = lane & 15;
    const int khi  = (lane >> 4) & 1;

    FragBF af[4], xf[4];
    {
        const unsigned short* ab = aggbf + (size_t)(r0 + nidx) * DIM;
        const unsigned short* xb = xbf   + (size_t)(r0 + nidx) * DIM;
        #pragma unroll
        for (int c = 0; c < 4; ++c) {
            af[c].q[0] = *(const uint4*)(ab + c * 32 + khi * 8);
            af[c].q[1] = *(const uint4*)(ab + c * 32 + 16 + khi * 8);
            xf[c].q[0] = *(const uint4*)(xb + c * 32 + khi * 8);
            xf[c].q[1] = *(const uint4*)(xb + c * 32 + 16 + khi * 8);
        }
    }

    #pragma unroll
    for (int dt = 0; dt < 8; ++dt) {
        const int ocol = dt * 16 + nidx;

        FragBF wfa[4], wfr[4];         // load all 8 weight frags, then chain
        #pragma unroll
        for (int c = 0; c < 4; ++c) {
            const unsigned short* wa = wbfA + (size_t)ocol * DIM + c * 32 + khi * 8;
            const unsigned short* wr = wbfR + (size_t)ocol * DIM + c * 32 + khi * 8;
            wfa[c].q[0] = *(const uint4*)(wa); wfa[c].q[1] = *(const uint4*)(wa + 16);
            wfr[c].q[0] = *(const uint4*)(wr); wfr[c].q[1] = *(const uint4*)(wr + 16);
        }

        v8f acc = {};
        #pragma unroll
        for (int c = 0; c < 4; ++c)    // agg @ W_att^T
            acc = __builtin_amdgcn_wmma_f32_16x16x32_bf16(
                    false, af[c].v, false, wfa[c].v, (short)0, acc, false, false);
        #pragma unroll
        for (int c = 0; c < 4; ++c)    // + x @ W_res^T
            acc = __builtin_amdgcn_wmma_f32_16x16x32_bf16(
                    false, xf[c].v, false, wfr[c].v, (short)0, acc, false, false);

        float s1 = 0.0f, s2 = 0.0f;
        float* pb = pre + (size_t)r0 * DIM;
        #pragma unroll
        for (int e = 0; e < 8; ++e) {
            const float v = acc[e];
            pb[(size_t)(e + khi * 8) * DIM + dt * 16 + nidx] = v;
            s1 += v; s2 += v * v;
        }
        s1 += __shfl_xor(s1, 16, 32);   // fold lane halves (same channel)
        s2 += __shfl_xor(s2, 16, 32);
        if (lane < 16) {
            atomicAdd(&stats[dt * 16 + lane], s1);
            atomicAdd(&stats[128 + dt * 16 + lane], s2);
        }
    }
}

// ---------------------------------------------------------------------------
// Kernel 4: BatchNorm (batch stats) + clamped-exp SELU
// ---------------------------------------------------------------------------
__global__ __launch_bounds__(256) void gal_finalize(
    const float* __restrict__ pre, const float* __restrict__ stats,
    const float* __restrict__ gamma, const float* __restrict__ beta,
    float* __restrict__ out)
{
    const int idx = blockIdx.x * 256 + threadIdx.x;
    const int c   = idx & (DIM - 1);
    const float invM = 1.0f / (float)(BATCH * NSEQ);
    const float mean = stats[c] * invM;
    const float var  = stats[128 + c] * invM - mean * mean;
    const float v  = (pre[idx] - mean) * rsqrtf(var + 1e-5f) * gamma[c] + beta[c];
    const float xc = fminf(fmaxf(v, -10.0f), 10.0f);
    out[idx] = SELU_SCALE * (v > 0.0f ? v : SELU_ALPHA * (__expf(xc) - 1.0f));
}

// ---------------------------------------------------------------------------
extern "C" void kernel_launch(void* const* d_in, const int* in_sizes, int n_in,
                              void* d_out, int out_size, void* d_ws, size_t ws_size,
                              hipStream_t stream) {
    const float* x     = (const float*)d_in[0];   // [4,2048,128]
    const float* wmap  = (const float*)d_in[1];   // [128]
    const float* Watt  = (const float*)d_in[2];   // [128,128]
    const float* Wres  = (const float*)d_in[3];   // [128,128]
    const float* gamma = (const float*)d_in[4];   // [128]
    const float* beta  = (const float*)d_in[5];   // [128]
    float* out = (float*)d_out;

    char* ws = (char*)d_ws;                       // needs ~10.3 MB
    unsigned short* qbf   = (unsigned short*)(ws);                 // 2 MB
    unsigned short* kbf   = (unsigned short*)(ws + (2u << 20));    // 2 MB
    unsigned short* aggbf = (unsigned short*)(ws + (4u << 20));    // 2 MB
    float* pre   = (float*)(ws + (6u << 20));                      // 4 MB
    float* stats = (float*)(ws + (10u << 20));                     // 1 KB
    unsigned short* wbfA = (unsigned short*)(ws + (10u << 20) + (64u << 10)); // 32 KB
    unsigned short* wbfR = (unsigned short*)(ws + (10u << 20) + (128u << 10));// 32 KB

    const int total = BATCH * NSEQ * DIM;         // 1,048,576

    gal_prep<<<total / 256, 256, 0, stream>>>(x, wmap, Watt, Wres, qbf, kbf, wbfA, wbfR, stats);
    gal_attn<<<BATCH * (NSEQ / ROWS_PB), WAVES * 32, 0, stream>>>(qbf, kbf, aggbf);
    gal_proj<<<(BATCH * NSEQ) / ROWS_PB, WAVES * 32, 0, stream>>>(aggbf, kbf, wbfA, wbfR, pre, stats);
    gal_finalize<<<total / 256, 256, 0, stream>>>(pre, stats, gamma, beta, out);
}
